// Net_63196148793447
// MI455X (gfx1250) — matrix operand order repaired and checked by reading, not compiled, not run on hardware
//
#include <hip/hip_runtime.h>

typedef __attribute__((ext_vector_type(16))) __bf16 v16bf;
typedef __attribute__((ext_vector_type(8)))  float  v8f;
typedef __attribute__((ext_vector_type(16))) unsigned short v16u;
typedef __attribute__((ext_vector_type(8)))  unsigned short v8u;

// f32 -> bf16 (RNE) via native conversion: lets the backend pick a hardware
// convert if gfx1250 has one, else it emits the canonical round sequence.
__device__ __forceinline__ unsigned short f2bf(float x) {
  __bf16 b = (__bf16)x;
  return __builtin_bit_cast(unsigned short, b);
}
__device__ __forceinline__ float bf2f(unsigned short s) {
  __bf16 b = __builtin_bit_cast(__bf16, s);
  return (float)b;
}

// ---------------------------------------------------------------- zero fill
__global__ void k_zero_f4(float4* p, long n4) {
  long i = (long)blockIdx.x * blockDim.x + threadIdx.x;
  if (i < n4) p[i] = make_float4(0.f, 0.f, 0.f, 0.f);
}

// ------------------------------------------------- pack weights to B-fragment
// Wpk layout per 64x64 matrix: element ((ct*2+kc)*32 + lane)*16 + j holds
// W[k][n] (bf16) with n = ct*16 + (lane&15), k = kc*32 + (lane>>4)*8 + (j>>3)*16 + (j&7).
// So each lane's v16bf B-fragment is one contiguous, 32B-aligned load.
__global__ void k_packW(const float* __restrict__ Wa, const float* __restrict__ Wb,
                        unsigned short* __restrict__ Wpk) {
  int t = blockIdx.x * blockDim.x + threadIdx.x;      // 8*4096 total
  int mi = t >> 12;
  int o  = t & 4095;
  int j    = o & 15;
  int lane = (o >> 4) & 31;
  int kc   = (o >> 9) & 1;
  int ct   = (o >> 10) & 3;
  int n = ct * 16 + (lane & 15);
  int k = kc * 32 + (lane >> 4) * 8 + (j >> 3) * 16 + (j & 7);
  const float* src = (mi < 4) ? (Wa + (size_t)mi * 4096) : (Wb + (size_t)(mi - 4) * 4096);
  Wpk[t] = f2bf(src[k * 64 + n]);
}

// ------------------------------------------------- edge aggregation, layer 0
__global__ void k_agg0(const int* __restrict__ ei, const float* __restrict__ x,
                       float* __restrict__ agg, int E) {
  int e = blockIdx.x * blockDim.x + threadIdx.x;
  if (e >= E) return;
  int s = ei[e];
  int d = ei[E + e];
  const float* xs = x + (size_t)s * 11;
  float* ad = agg + (size_t)d * 16;
  #pragma unroll
  for (int j = 0; j < 11; j++) atomicAdd(ad + j, xs[j]);
}

// ------------------------------------------------- edge aggregation, 64 feat
// thread = (edge, 8-feature chunk): 16B bf16 gather + 8 f32 atomic adds
__global__ void k_agg64(const int* __restrict__ ei, const unsigned short* __restrict__ h,
                        float* __restrict__ agg, int E) {
  int t = blockIdx.x * blockDim.x + threadIdx.x;
  if (t >= E * 8) return;
  int e = t >> 3;
  int c = (t & 7) * 8;
  int s = ei[e];
  int d = ei[E + e];
  v8u v = *(const v8u*)(h + (size_t)s * 64 + c);
  float* ap = agg + (size_t)d * 64 + c;
  #pragma unroll
  for (int j = 0; j < 8; j++) atomicAdd(ap + j, bf2f(v[j]));
}

// ------------------------------------------------- layer-0 MLP (11->64->64)
__global__ __launch_bounds__(64) void k_mlp0(
    const float* __restrict__ x, const float* __restrict__ agg,
    const float* __restrict__ W1a, const float* __restrict__ b1a,
    const float* __restrict__ W1b, const float* __restrict__ b1b,
    unsigned short* __restrict__ u, float* __restrict__ bnstat, int N) {
  __shared__ float a[16];
  __shared__ float t[64];
  int f = threadIdx.x;
  int i0 = blockIdx.x * 64;
  int iend = i0 + 64; if (iend > N) iend = N;
  float s = 0.f, q = 0.f;
  for (int i = i0; i < iend; i++) {
    if (f < 11) a[f] = x[(size_t)i * 11 + f] + agg[(size_t)i * 16 + f];
    __syncthreads();
    float tv = b1a[f];
    #pragma unroll
    for (int j = 0; j < 11; j++) tv += a[j] * W1a[j * 64 + f];
    tv = fmaxf(tv, 0.f);
    t[f] = tv;
    __syncthreads();
    float uv = b1b[f];
    for (int j = 0; j < 64; j++) uv += t[j] * W1b[j * 64 + f];
    uv = fmaxf(uv, 0.f);
    u[(size_t)i * 64 + f] = f2bf(uv);
    s += uv; q += uv * uv;
    __syncthreads();
  }
  atomicAdd(&bnstat[f], s);
  atomicAdd(&bnstat[64 + f], q);
}

// ------------------------------------------------- WMMA GIN MLP (64->64->64)
// 256 threads = 8 waves; block covers 128 padded rows; wave owns 16 rows.
// u = relu(relu((h+agg) @ Wa + ba) @ Wb + bb); fused BN sum/sumsq stats.
__global__ __launch_bounds__(256) void k_mlp_wmma(
    const unsigned short* __restrict__ h, const float* __restrict__ agg,
    const unsigned short* __restrict__ WpkA, const unsigned short* __restrict__ WpkB,
    const float* __restrict__ ba, const float* __restrict__ bb,
    unsigned short* __restrict__ u, float* __restrict__ bnstat, int N) {
  __shared__ unsigned short tl[128 * 64];  // t matrix staging (wave-private rows)
  __shared__ float st[128];
  int tid = threadIdx.x;
  if (tid < 128) st[tid] = 0.f;
  __syncthreads();

  int w = tid >> 5, lane = tid & 31;
  int lr = lane & 15, lh = lane >> 4;
  int m0 = blockIdx.x * 128 + w * 16;
  int row = m0 + lr;
  const unsigned short* hr = h + (size_t)row * 64;
  const float* ar = agg + (size_t)row * 64;

  // A fragments: lane = row lr; lh picks K interleave {0..7,16..23} / {8..15,24..31}
  v16u A[2];
  #pragma unroll
  for (int kc = 0; kc < 2; kc++) {
    int k0 = kc * 32 + lh * 8;
    v8u h0 = *(const v8u*)(hr + k0);
    v8u h1 = *(const v8u*)(hr + k0 + 16);
    #pragma unroll
    for (int j = 0; j < 8; j++) {
      A[kc][j]     = f2bf(bf2f(h0[j]) + ar[k0 + j]);       // v_fma_mix_f32_bf16
      A[kc][j + 8] = f2bf(bf2f(h1[j]) + ar[k0 + 16 + j]);
    }
  }

  // GEMM1: t = relu((h+agg) @ Wa + ba), written to LDS in row-major
  #pragma unroll
  for (int ct = 0; ct < 4; ct++) {
    v8f acc = {};
    #pragma unroll
    for (int kc = 0; kc < 2; kc++) {
      v16u Bu = *(const v16u*)(WpkA + (((ct * 2 + kc) * 32 + lane) << 4));
      acc = __builtin_amdgcn_wmma_f32_16x16x32_bf16(
          false, __builtin_bit_cast(v16bf, A[kc]),
          false, __builtin_bit_cast(v16bf, Bu),
          (short)0, acc, false, false);
    }
    int n = ct * 16 + lr;
    float bias = ba[n];
    #pragma unroll
    for (int r = 0; r < 8; r++) {
      float v = acc[r] + bias;
      v = v > 0.f ? v : 0.f;
      tl[(w * 16 + lh * 8 + r) * 64 + n] = f2bf(v);  // wave-private rows: no barrier
    }
  }

  // reload t as A fragments (D column-layout -> A row-layout via LDS)
  v16u A2[2];
  const unsigned short* trow = &tl[(w * 16 + lr) * 64];
  #pragma unroll
  for (int kc = 0; kc < 2; kc++) {
    int k0 = kc * 32 + lh * 8;
    v8u t0 = *(const v8u*)(trow + k0);
    v8u t1 = *(const v8u*)(trow + k0 + 16);
    #pragma unroll
    for (int j = 0; j < 8; j++) { A2[kc][j] = t0[j]; A2[kc][j + 8] = t1[j]; }
  }

  // GEMM2: u = relu(t @ Wb + bb) + BN statistics
  #pragma unroll
  for (int ct = 0; ct < 4; ct++) {
    v8f acc = {};
    #pragma unroll
    for (int kc = 0; kc < 2; kc++) {
      v16u Bu = *(const v16u*)(WpkB + (((ct * 2 + kc) * 32 + lane) << 4));
      acc = __builtin_amdgcn_wmma_f32_16x16x32_bf16(
          false, __builtin_bit_cast(v16bf, A2[kc]),
          false, __builtin_bit_cast(v16bf, Bu),
          (short)0, acc, false, false);
    }
    int n = ct * 16 + lr;
    float bias = bb[n];
    float s = 0.f, q = 0.f;
    #pragma unroll
    for (int r = 0; r < 8; r++) {
      int m = m0 + lh * 8 + r;
      float v = acc[r] + bias;
      v = v > 0.f ? v : 0.f;
      u[(size_t)m * 64 + n] = f2bf(v);
      if (m < N) { s += v; q += v * v; }   // exclude pad rows from BN stats
    }
    atomicAdd(&st[n], s);
    atomicAdd(&st[64 + n], q);
  }
  __syncthreads();
  if (tid < 128) atomicAdd(&bnstat[tid], st[tid]);
}

// ------------------------------------------------- BN finalize -> scale/shift
__global__ __launch_bounds__(64) void k_bnfin(const float* __restrict__ stat,
                                              const float* __restrict__ gamma,
                                              const float* __restrict__ beta,
                                              float* __restrict__ sc, float* __restrict__ sh,
                                              float invN) {
  int f = threadIdx.x;
  float mean = stat[f] * invN;
  float var = stat[64 + f] * invN - mean * mean;
  float inv = rsqrtf(var + 1e-5f);
  float s = gamma[f] * inv;
  sc[f] = s;
  sh[f] = beta[f] - mean * s;
}

// ------------------------------------------------- BN apply + graph pooling
__global__ __launch_bounds__(64) void k_bnpool(
    const unsigned short* __restrict__ u, const float* __restrict__ sc,
    const float* __restrict__ sh, const int* __restrict__ batch,
    unsigned short* __restrict__ hout, float* __restrict__ pooled, int off, int N) {
  int f = threadIdx.x;
  int i0 = blockIdx.x * 256;
  int iend = i0 + 256; if (iend > N) iend = N;
  float s = sc[f], b = sh[f];
  float acc = 0.f;
  int g = batch[i0];
  for (int i = i0; i < iend; i++) {
    float v = bf2f(u[(size_t)i * 64 + f]) * s + b;
    hout[(size_t)i * 64 + f] = f2bf(v);
    int gi = batch[i];
    if (gi != g) {                      // batch is sorted: flush on graph change
      atomicAdd(&pooled[(size_t)g * 320 + off + f], acc);
      acc = 0.f; g = gi;
    }
    acc += v;
  }
  atomicAdd(&pooled[(size_t)g * 320 + off + f], acc);
}

// ------------------------------------------------- readout head
__global__ __launch_bounds__(64) void k_head(
    const float* __restrict__ pooled, const float* __restrict__ W1,
    const float* __restrict__ b1, const float* __restrict__ W2,
    const float* __restrict__ b2, float* __restrict__ out) {
  __shared__ float z[320];
  __shared__ float red[64];
  int g = blockIdx.x, f = threadIdx.x;
  for (int k = f; k < 320; k += 64) z[k] = pooled[(size_t)g * 320 + k];
  __syncthreads();
  float a = b1[f];
  for (int k = 0; k < 320; k++) a += z[k] * W1[(size_t)k * 64 + f];
  a = fmaxf(a, 0.f);
  red[f] = a * W2[f];
  __syncthreads();
  if (f == 0) {
    float s = b2[0];
    #pragma unroll
    for (int k = 0; k < 64; k++) s += red[k];
    out[g] = s;
  }
}

// ================================================================ launcher
extern "C" void kernel_launch(void* const* d_in, const int* in_sizes, int n_in,
                              void* d_out, int out_size, void* d_ws, size_t ws_size,
                              hipStream_t stream) {
  const float* x     = (const float*)d_in[0];
  const int*   ei    = (const int*)d_in[1];
  const int*   batch = (const int*)d_in[2];
  const float* W1a   = (const float*)d_in[3];
  const float* b1a   = (const float*)d_in[4];
  const float* W1b   = (const float*)d_in[5];
  const float* b1b   = (const float*)d_in[6];
  const float* Wa    = (const float*)d_in[7];
  const float* ba    = (const float*)d_in[8];
  const float* Wb    = (const float*)d_in[9];
  const float* bb    = (const float*)d_in[10];
  const float* gamma = (const float*)d_in[11];
  const float* beta  = (const float*)d_in[12];
  const float* fc1W  = (const float*)d_in[13];
  const float* fc1b  = (const float*)d_in[14];
  const float* fc2W  = (const float*)d_in[15];
  const float* fc2b  = (const float*)d_in[16];

  const int N = in_sizes[0] / 11;
  const int E = in_sizes[1] / 2;
  const int G = out_size;
  const int NPAD = ((N + 127) / 128) * 128;

  // workspace carve (all 256B aligned)
  char* ws = (char*)d_ws;
  size_t off = 0;
  auto carve = [&](size_t bytes) -> void* {
    void* p = ws + off;
    off = (off + bytes + 255) & ~(size_t)255;
    return p;
  };
  unsigned short* hA   = (unsigned short*)carve((size_t)NPAD * 64 * 2);
  unsigned short* hB   = (unsigned short*)carve((size_t)NPAD * 64 * 2);
  unsigned short* ubuf = (unsigned short*)carve((size_t)NPAD * 64 * 2);
  float* agg    = (float*)carve((size_t)NPAD * 64 * 4);   // layer0 reuses w/ stride 16
  unsigned short* Wpk = (unsigned short*)carve(8 * 4096 * 2);
  float* bnstat = (float*)carve(128 * 4);
  float* bnsc   = (float*)carve(64 * 4);
  float* bnsh   = (float*)carve(64 * 4);
  float* pooled = (float*)carve((size_t)G * 320 * 4);

  auto zero = [&](void* p, long floats) {
    long n4 = floats / 4;
    k_zero_f4<<<dim3((unsigned)((n4 + 255) / 256)), dim3(256), 0, stream>>>((float4*)p, n4);
  };

  // one-time-per-launch prep
  k_packW<<<dim3(128), dim3(256), 0, stream>>>(Wa, Wb, Wpk);
  zero(pooled, (long)G * 320);
  zero(hA + (size_t)N * 64, (long)(NPAD - N) * 64 / 2);   // bf16 pad rows -> f32 count
  zero(hB + (size_t)N * 64, (long)(NPAD - N) * 64 / 2);

  // ---------------- layer 0 (11 -> 64 -> 64) ----------------
  zero(agg, (long)NPAD * 16);
  k_agg0<<<dim3((unsigned)((E + 255) / 256)), dim3(256), 0, stream>>>(ei, x, agg, E);
  zero(bnstat, 128);
  k_mlp0<<<dim3((unsigned)((N + 63) / 64)), dim3(64), 0, stream>>>(
      x, agg, W1a, b1a, W1b, b1b, ubuf, bnstat, N);
  k_bnfin<<<dim3(1), dim3(64), 0, stream>>>(bnstat, gamma, beta, bnsc, bnsh, 1.0f / N);
  k_bnpool<<<dim3((unsigned)((N + 255) / 256)), dim3(64), 0, stream>>>(
      ubuf, bnsc, bnsh, batch, hA, pooled, 0, N);

  // ---------------- layers 1..4 (WMMA MLPs) ----------------
  unsigned short* hcur = hA;
  unsigned short* hnext = hB;
  for (int l = 0; l < 4; l++) {
    zero(agg, (long)NPAD * 64);
    k_agg64<<<dim3((unsigned)(((long)E * 8 + 255) / 256)), dim3(256), 0, stream>>>(
        ei, hcur, agg, E);
    zero(bnstat, 128);
    k_mlp_wmma<<<dim3((unsigned)(NPAD / 128)), dim3(256), 0, stream>>>(
        hcur, agg, Wpk + (size_t)l * 4096, Wpk + (size_t)(4 + l) * 4096,
        ba + (size_t)l * 64, bb + (size_t)l * 64, ubuf, bnstat, N);
    k_bnfin<<<dim3(1), dim3(64), 0, stream>>>(
        bnstat, gamma + (size_t)(l + 1) * 64, beta + (size_t)(l + 1) * 64,
        bnsc, bnsh, 1.0f / N);
    k_bnpool<<<dim3((unsigned)((N + 255) / 256)), dim3(64), 0, stream>>>(
        ubuf, bnsc, bnsh, batch, hnext, pooled, (l + 1) * 64, N);
    unsigned short* tmp = hcur; hcur = hnext; hnext = tmp;
  }

  // ---------------- readout head ----------------
  k_head<<<dim3((unsigned)G), dim3(64), 0, stream>>>(pooled, fc1W, fc1b, fc2W, fc2b,
                                                     (float*)d_out);
}
